// NNUENet_14439680049708
// MI455X (gfx1250) — matrix operand order repaired and checked by reading, not compiled, not run on hardware
//
#include <hip/hip_runtime.h>
#include <hip/hip_bf16.h>
#include <math.h>

typedef _Float16 v16h __attribute__((ext_vector_type(16)));
typedef _Float16 v8h  __attribute__((ext_vector_type(8)));
typedef float    v8f  __attribute__((ext_vector_type(8)));
typedef float    v4f  __attribute__((ext_vector_type(4)));

#define FEATS   41024
#define HID     256      // per-side hidden
#define BAG     32
#define BATCH   8192
#define TILE_M  16       // batch rows per block (WMMA M)
#define K1      512      // 2*HID, fc1 reduction dim
#define XS      520      // f16 stride of x tile (512 + 8 pad, multiple of 8)
#define HS      40       // f16 stride of h1 tile (32 + 8 pad)
#define H2S     33       // f32 stride of h2 tile

__global__ __launch_bounds__(256) void nnue_fused_kernel(
    const int*   __restrict__ us,
    const int*   __restrict__ them,
    const float* __restrict__ emb,
    const float* __restrict__ fc1_w, const float* __restrict__ fc1_b,
    const float* __restrict__ fc2_w, const float* __restrict__ fc2_b,
    const float* __restrict__ fc3_w, const float* __restrict__ fc3_b,
    float* __restrict__ out)
{
    __shared__ __align__(16) _Float16 xh [TILE_M * XS];   // concat(acc_us, acc_them) as f16
    __shared__ __align__(16) _Float16 h1h[TILE_M * HS];   // fc1 activations as f16
    __shared__ __align__(16) float    h2f[TILE_M * H2S];  // fc2 activations as f32

    const int lane = threadIdx.x & 31;
    const int wave = threadIdx.x >> 5;     // 0..7
    const int tile = blockIdx.x;           // 0..511

    // ---------------- Phase 1: embedding-bag gather+sum (all 8 waves) -------------
    // 32 (row,side) pairs per block; wave owns 4 of them.
    #pragma unroll
    for (int p = 0; p < 4; ++p) {
        const int gp   = wave * 4 + p;       // 0..31
        const int r    = gp >> 1;            // tile row 0..15
        const int s    = gp & 1;             // 0 = us, 1 = them
        const int grow = tile * TILE_M + r;  // global batch row
        const int* bags = (s ? them : us) + grow * BAG;
        const int idxv  = bags[lane];        // coalesced: 32 indices for this bag

        v4f acc0 = {0.f, 0.f, 0.f, 0.f};
        v4f acc1 = {0.f, 0.f, 0.f, 0.f};
        #pragma unroll 4
        for (int j = 0; j < BAG; ++j) {
            const int ridx = __shfl(idxv, j, 32);          // wave-uniform row index
            const v4f* rowp = (const v4f*)(emb + (size_t)ridx * HID);
            acc0 += rowp[lane];        // floats [lane*4,     lane*4+4)
            acc1 += rowp[lane + 32];   // floats [128+lane*4, 128+lane*4+4)
        }
        // store as f16 into x tile: x[r][s*256 + c]
        _Float16* dst = xh + r * XS + s * HID + lane * 4;
        #pragma unroll
        for (int i = 0; i < 4; ++i) {
            dst[i]       = (_Float16)acc0[i];
            dst[128 + i] = (_Float16)acc1[i];
        }
    }
    __syncthreads();

    // ---------------- Phase 2: fc1 = clip01(x @ fc1_w^T + b1), WMMA f16 ----------
    // waves 0,1 each own one 16-col N-tile; K = 512 -> 16 wmma steps.
    if (wave < 2) {
        const int half = lane >> 4;        // selects K sub-range per ISA layout
        const int m    = lane & 15;        // A-matrix row
        const int nn   = wave * 16 + (lane & 15);   // global output column 0..31

        v8f c = {};
        #pragma unroll
        for (int kt = 0; kt < 16; ++kt) {
            const int k0 = kt * 32;
            // A fragment: a[e] = x[m][k0 + (e<8 ? half*8+e : 16+half*8+(e-8))]
            const _Float16* ap = xh + m * XS + k0 + half * 8;
            v8h lo = *(const v8h*)ap;          // K = k0 + half*8 .. +7
            v8h hi = *(const v8h*)(ap + 16);   // K = k0 + 16 + half*8 .. +7
            v16h a = __builtin_shufflevector(lo, hi,
                        0,1,2,3,4,5,6,7,8,9,10,11,12,13,14,15);
            // B fragment: b[e] = fc1_w[nn][k0 + (e<8 ? half*8+e : 16+half*8+(e-8))], f32 -> f16 on the fly
            const float* wp = fc1_w + nn * K1 + k0 + half * 8;
            v16h b;
            #pragma unroll
            for (int e = 0; e < 16; ++e) b[e] = (_Float16)wp[e < 8 ? e : e + 8];

            c = __builtin_amdgcn_wmma_f32_16x16x32_f16(
                    false, a, false, b, (short)0, c, false, false);
        }
        // bias + clip01; element v holds row m = half*8+v, column nn
        const float bias = fc1_b[nn];
        #pragma unroll
        for (int v = 0; v < 8; ++v) {
            float val = c[v] + bias;
            val = fminf(fmaxf(val, 0.0f), 1.0f);
            h1h[(half * 8 + v) * HS + nn] = (_Float16)val;
        }
    }
    __syncthreads();

    // ---------------- Phase 3: fc2 = clip01(h1 @ fc2_w^T + b2), single WMMA ------
    if (wave < 2) {
        const int half = lane >> 4;
        const int m    = lane & 15;
        const int nn   = wave * 16 + (lane & 15);

        const _Float16* ap = h1h + m * HS + half * 8;
        v8h lo = *(const v8h*)ap;
        v8h hi = *(const v8h*)(ap + 16);
        v16h a = __builtin_shufflevector(lo, hi,
                    0,1,2,3,4,5,6,7,8,9,10,11,12,13,14,15);

        const float* wp = fc2_w + nn * 32 + half * 8;
        v16h b;
        #pragma unroll
        for (int e = 0; e < 16; ++e) b[e] = (_Float16)wp[e < 8 ? e : e + 8];

        v8f c = {};
        c = __builtin_amdgcn_wmma_f32_16x16x32_f16(
                false, a, false, b, (short)0, c, false, false);

        const float bias = fc2_b[nn];
        #pragma unroll
        for (int v = 0; v < 8; ++v) {
            float val = c[v] + bias;
            val = fminf(fmaxf(val, 0.0f), 1.0f);
            h2f[(half * 8 + v) * H2S + nn] = val;
        }
    }
    __syncthreads();

    // ---------------- Phase 4: fc3 + tanh (one lane per batch row) ---------------
    if (threadIdx.x < TILE_M) {
        const int r = threadIdx.x;
        float acc = fc3_b[0];
        #pragma unroll
        for (int n = 0; n < 32; ++n)
            acc += h2f[r * H2S + n] * fc3_w[n];
        out[tile * TILE_M + r] = tanhf(acc);
    }
}

extern "C" void kernel_launch(void* const* d_in, const int* in_sizes, int n_in,
                              void* d_out, int out_size, void* d_ws, size_t ws_size,
                              hipStream_t stream) {
    const int*   us    = (const int*)  d_in[0];
    const int*   them  = (const int*)  d_in[1];
    const float* emb   = (const float*)d_in[2];
    const float* fc1_w = (const float*)d_in[3];
    const float* fc1_b = (const float*)d_in[4];
    const float* fc2_w = (const float*)d_in[5];
    const float* fc2_b = (const float*)d_in[6];
    const float* fc3_w = (const float*)d_in[7];
    const float* fc3_b = (const float*)d_in[8];
    float* out = (float*)d_out;

    dim3 grid(BATCH / TILE_M);   // 512 blocks
    dim3 block(256);             // 8 wave32
    hipLaunchKernelGGL(nnue_fused_kernel, grid, block, 0, stream,
                       us, them, emb, fc1_w, fc1_b, fc2_w, fc2_b, fc3_w, fc3_b, out);
}
